// NonLocalBlockRef_36524401885627
// MI455X (gfx1250) — matrix operand order repaired and checked by reading, not compiled, not run on hardware
//
#include <hip/hip_runtime.h>
#include <hip/hip_bf16.h>

// ---------------------------------------------------------------------------
// NonLocalBlock for MI455X (gfx1250), bf16 WMMA path, flash-style 2-pass
// softmax over the query axis (axis=1) to avoid materializing the 340MB
// scores tensor.
//
// Dims (fixed by reference): B=4, C=128, INTER=64, H=W=96, HW=9216, N=4608.
// ---------------------------------------------------------------------------

#define NB    4
#define CC    128
#define NI    64
#define HWp   9216
#define NN    4608

typedef __attribute__((ext_vector_type(16))) __bf16 v16bf;
typedef __attribute__((ext_vector_type(8)))  __bf16 v8bf;
typedef __attribute__((ext_vector_type(8)))  float  v8f;

__device__ __forceinline__ v8f wmma_bf16(v16bf a, v16bf b, v8f c) {
  // D = A(16x32 bf16) * B(32x16 bf16) + C(16x16 f32)
  return __builtin_amdgcn_wmma_f32_16x16x32_bf16(
      false, a, false, b, (short)0, c, false, false);
}

__device__ __forceinline__ v16bf concat16(v8bf lo, v8bf hi) {
  v16bf r;
#pragma unroll
  for (int e = 0; e < 8; ++e) { r[e] = lo[e]; r[e + 8] = hi[e]; }
  return r;
}

// B-operand: lane holds col N = lane%16, k = kbase + e + 16*(lane>>4); caller
// passes pointer at the start of the 16 contiguous bf16 k-values.
__device__ __forceinline__ v16bf load16(const __bf16* p) {
  return concat16(*(const v8bf*)p, *(const v8bf*)(p + 8));
}

// A-operand: lane holds row M = lane%16; k=kbase+16*(lane>>4)+e.
// Two contiguous 16B chunks from a row-major A row.
__device__ __forceinline__ v16bf load_a32(const __bf16* row, int kbase, int lane) {
  const int k0 = kbase + 16 * (lane >> 4);
  return concat16(*(const v8bf*)(row + k0), *(const v8bf*)(row + k0 + 8));
}

// ---------------------------------------------------------------------------
// Kernel 1: fused 1x1-conv projections theta/phi/g, fp32 -> bf16 WMMA.
//   theta/phi stored with the "buggy reshape" applied AND transposed to
//   [n, ci] (rows contiguous over ci=128) so attention A/B fetches are linear.
//   g stored in natural flat [64,9216] layout == reshaped [128,4608].
// grid = (288, B*3), block = 256 (8 waves: 4 i-tiles x 2 p-subtiles of 16)
// ---------------------------------------------------------------------------
__global__ __launch_bounds__(256) void proj_kernel(
    const float* __restrict__ x1, const float* __restrict__ x2,
    const float* __restrict__ w_theta, const float* __restrict__ w_phi,
    const float* __restrict__ w_g,
    __bf16* __restrict__ theta_t, __bf16* __restrict__ phi_t,
    __bf16* __restrict__ g_flat) {
  const int ptile = blockIdx.x;            // 32 pixels per workgroup
  const int b = blockIdx.y / 3;
  const int proj = blockIdx.y % 3;         // 0=theta(x1) 1=phi(x2) 2=g(x2)
  const float* x = (proj == 0) ? x1 : x2;
  const float* w = (proj == 0) ? w_theta : (proj == 1 ? w_phi : w_g);
  const int p0 = ptile * 32;

  // Stage x tile transposed: xT[p][c] in bf16, padded pitch to spread banks.
  __shared__ __bf16 xT[32][136];
  {
    const int t = threadIdx.x;
    const int p = t & 31;
    for (int c = (t >> 5); c < CC; c += 8)
      xT[p][c] = (__bf16)x[((size_t)b * CC + c) * HWp + p0 + p];
  }
  __syncthreads();

  const int wave = threadIdx.x >> 5;
  const int lane = threadIdx.x & 31;
  const int itile = wave >> 1;             // 0..3 -> rows of INTER=64
  const int psub = wave & 1;               // 0..1 -> 16-pixel subtile

  // Preload A = weight rows (fp32 -> bf16), K=128 -> 4 chunks of 32
  v16bf A[4];
  {
    const float* wr = w + (itile * 16 + (lane & 15)) * CC;
#pragma unroll
    for (int kc = 0; kc < 4; ++kc) {
      const int k0 = kc * 32 + 16 * (lane >> 4);
      v16bf a;
#pragma unroll
      for (int e = 0; e < 8; ++e) {
        a[e] = (__bf16)wr[k0 + e];
        a[e + 8] = (__bf16)wr[k0 + 8 + e];
      }
      A[kc] = a;
    }
  }

  const int pl = psub * 16 + (lane & 15);  // local pixel col for this lane
  v8f acc = {};
#pragma unroll
  for (int kc = 0; kc < 4; ++kc) {
    const int k0 = kc * 32 + 16 * (lane >> 4);
    v16bf bm = concat16(*(const v8bf*)&xT[pl][k0], *(const v8bf*)&xT[pl][k0 + 8]);
    acc = wmma_bf16(A[kc], bm, acc);
  }

  // Scatter D (lane: col p fixed, row i = itile*16 + r + 8*(lane>>4))
  const int pg = p0 + pl;
#pragma unroll
  for (int r = 0; r < 8; ++r) {
    const int i = itile * 16 + 8 * (lane >> 4) + r;
    const __bf16 v = (__bf16)acc[r];
    if (proj == 2) {
      g_flat[(size_t)b * (NI * HWp) + (size_t)i * HWp + pg] = v;
    } else {
      const int ci = 2 * i + (pg >= NN ? 1 : 0);  // buggy-reshape channel
      const int n = (pg < NN) ? pg : pg - NN;
      __bf16* dst = (proj == 0 ? theta_t : phi_t);
      dst[(size_t)b * (NN * CC) + (size_t)n * CC + ci] = v;
    }
  }
}

// ---------------------------------------------------------------------------
// Kernel 2: pass 1 — per-column softmax stats over the query axis n.
//   M[m] = max_n s[n,m],  rZ[m] = 1 / sum_n exp(s[n,m]-M[m])  (online)
// grid = (NN/128, B), block = 256 (each wave owns 16 columns, loops all n)
// ---------------------------------------------------------------------------
__global__ __launch_bounds__(256) void stats_kernel(
    const __bf16* __restrict__ theta_t, const __bf16* __restrict__ phi_t,
    float* __restrict__ Mbuf, float* __restrict__ rZbuf) {
  const int b = blockIdx.y;
  const int wave = threadIdx.x >> 5, lane = threadIdx.x & 31;
  const int m0 = blockIdx.x * 128 + wave * 16;
  const __bf16* th = theta_t + (size_t)b * NN * CC;
  const __bf16* ph = phi_t + (size_t)b * NN * CC;

  // B = phi rows (fixed per wave for the entire n loop)
  v16bf Bm[4];
  {
    const __bf16* prow = ph + (size_t)(m0 + (lane & 15)) * CC;
#pragma unroll
    for (int kc = 0; kc < 4; ++kc)
      Bm[kc] = load16(prow + kc * 32 + 16 * (lane >> 4));
  }

  float mx = -3.0e38f, sm = 0.0f;
  for (int n0 = 0; n0 < NN; n0 += 16) {
    const __bf16* arow = th + (size_t)(n0 + (lane & 15)) * CC;
    v8f s = {};
#pragma unroll
    for (int kc = 0; kc < 4; ++kc)
      s = wmma_bf16(load_a32(arow, kc * 32, lane), Bm[kc], s);
    // per-lane online softmax update over the 8 rows this lane holds
    float tmax = s[0];
#pragma unroll
    for (int r = 1; r < 8; ++r) tmax = fmaxf(tmax, s[r]);
    const float nmx = fmaxf(mx, tmax);
    float ts = 0.0f;
#pragma unroll
    for (int r = 0; r < 8; ++r) ts += __expf(s[r] - nmx);
    sm = sm * __expf(mx - nmx) + ts;
    mx = nmx;
  }
  // lanes L and L^16 hold the same column -> combine
  const float omx = __shfl_xor(mx, 16, 32);
  const float osm = __shfl_xor(sm, 16, 32);
  const float cmx = fmaxf(mx, omx);
  const float csm = sm * __expf(mx - cmx) + osm * __expf(omx - cmx);
  if (lane < 16) {
    Mbuf[b * NN + m0 + lane] = cmx;
    rZbuf[b * NN + m0 + lane] = 1.0f / csm;
  }
}

// ---------------------------------------------------------------------------
// Kernel 3: pass 2 — recompute score tiles, p=exp(s-M)*rZ, y += p * g.
//   Each wave: 16 query rows, full ci=128 accumulator (8 octets), loops m.
//   D->A transpose of p goes through a per-wave LDS tile (DS in-order).
// grid = (NN/128, B), block = 256
// ---------------------------------------------------------------------------
__global__ __launch_bounds__(256) void attn_kernel(
    const __bf16* __restrict__ theta_t, const __bf16* __restrict__ phi_t,
    const __bf16* __restrict__ g_flat, const float* __restrict__ Mbuf,
    const float* __restrict__ rZbuf, __bf16* __restrict__ y2) {
  const int b = blockIdx.y;
  const int wave = threadIdx.x >> 5, lane = threadIdx.x & 31;
  const int n0 = blockIdx.x * 128 + wave * 16;
  const __bf16* th = theta_t + (size_t)b * NN * CC;
  const __bf16* ph = phi_t + (size_t)b * NN * CC;
  const __bf16* gg = g_flat + (size_t)b * (NI * HWp);  // viewed [128][4608]
  const float* Mb = Mbuf + b * NN;
  const float* Rb = rZbuf + b * NN;

  __shared__ __bf16 pt[8][16][40];  // per-wave 16x32 p tile, padded pitch

  // Preload A = theta rows (fixed for entire m loop)
  v16bf Ath[4];
  {
    const __bf16* arow = th + (size_t)(n0 + (lane & 15)) * CC;
#pragma unroll
    for (int kc = 0; kc < 4; ++kc) Ath[kc] = load_a32(arow, kc * 32, lane);
  }

  v8f y[8];
#pragma unroll
  for (int t = 0; t < 8; ++t) y[t] = v8f{};

  for (int m0 = 0; m0 < NN; m0 += 32) {
#pragma unroll
    for (int half = 0; half < 2; ++half) {
      const int mb = m0 + half * 16;
      const __bf16* prow = ph + (size_t)(mb + (lane & 15)) * CC;
      v8f s = {};
#pragma unroll
      for (int kc = 0; kc < 4; ++kc)
        s = wmma_bf16(Ath[kc], load16(prow + kc * 32 + 16 * (lane >> 4)), s);
      const int mcol = mb + (lane & 15);
      const float Mv = Mb[mcol], Rv = Rb[mcol];
#pragma unroll
      for (int r = 0; r < 8; ++r)
        pt[wave][r + 8 * (lane >> 4)][half * 16 + (lane & 15)] =
            (__bf16)(__expf(s[r] - Mv) * Rv);
    }
    asm volatile("" ::: "memory");  // keep DS store->load program order
    // A = p tile from LDS (within-wave transpose; DS ops in-order per wave)
    v16bf Ap;
    {
      const __bf16* prow = &pt[wave][lane & 15][0];
      const int k0 = 16 * (lane >> 4);
      Ap = concat16(*(const v8bf*)(prow + k0), *(const v8bf*)(prow + k0 + 8));
    }
    // y[n, ci] += p[n, m] * g[m, ci] over all 8 ci-tiles
#pragma unroll
    for (int t = 0; t < 8; ++t) {
      const int ci = t * 16 + (lane & 15);
      v16bf bg = load16(gg + (size_t)ci * NN + m0 + 16 * (lane >> 4));
      y[t] = wmma_bf16(Ap, bg, y[t]);
    }
  }

  // Store y pre-transposed for the mask GEMM: y2[p][i] with
  // p = n + (ci&1)*NN, i = ci>>1 (inverse of the reference reshape).
#pragma unroll
  for (int t = 0; t < 8; ++t) {
    const int ci = t * 16 + (lane & 15);
#pragma unroll
    for (int r = 0; r < 8; ++r) {
      const int n = n0 + r + 8 * (lane >> 4);
      const int pp = n + (ci & 1) * NN;
      y2[(size_t)b * (HWp * NI) + (size_t)pp * NI + (ci >> 1)] = (__bf16)y[t][r];
    }
  }
}

// ---------------------------------------------------------------------------
// Kernel 4: mask = w_mask @ y  (+ x1 residual), fp32 output.
// grid = (HWp/16, B), block = 256 (8 waves = 8 co-tiles covering C=128)
// ---------------------------------------------------------------------------
__global__ __launch_bounds__(256) void mask_kernel(
    const float* __restrict__ w_mask, const __bf16* __restrict__ y2,
    const float* __restrict__ x1, float* __restrict__ out) {
  const int b = blockIdx.y;
  const int p0 = blockIdx.x * 16;
  const int wave = threadIdx.x >> 5, lane = threadIdx.x & 31;

  // A = w_mask rows [co, i] fp32 -> bf16, K=64 -> 2 chunks
  v16bf A[2];
  {
    const float* wr = w_mask + (wave * 16 + (lane & 15)) * NI;
#pragma unroll
    for (int kc = 0; kc < 2; ++kc) {
      const int k0 = kc * 32 + 16 * (lane >> 4);
      v16bf a;
#pragma unroll
      for (int e = 0; e < 8; ++e) {
        a[e] = (__bf16)wr[k0 + e];
        a[e + 8] = (__bf16)wr[k0 + 8 + e];
      }
      A[kc] = a;
    }
  }

  const __bf16* yb = y2 + (size_t)b * HWp * NI;
  const int pl = p0 + (lane & 15);
  v8f acc = {};
#pragma unroll
  for (int kc = 0; kc < 2; ++kc) {
    v16bf bm = load16(yb + (size_t)pl * NI + kc * 32 + 16 * (lane >> 4));
    acc = wmma_bf16(A[kc], bm, acc);
  }
#pragma unroll
  for (int r = 0; r < 8; ++r) {
    const int co = wave * 16 + r + 8 * (lane >> 4);
    const size_t idx = ((size_t)b * CC + co) * HWp + pl;
    out[idx] = acc[r] + x1[idx];
  }
}

// ---------------------------------------------------------------------------
extern "C" void kernel_launch(void* const* d_in, const int* in_sizes, int n_in,
                              void* d_out, int out_size, void* d_ws,
                              size_t ws_size, hipStream_t stream) {
  const float* x1 = (const float*)d_in[0];
  const float* x2 = (const float*)d_in[1];
  const float* w_theta = (const float*)d_in[2];
  const float* w_phi = (const float*)d_in[3];
  const float* w_g = (const float*)d_in[4];
  const float* w_mask = (const float*)d_in[5];
  float* out = (float*)d_out;

  // Workspace carve-out (~18.2 MB total)
  char* ws = (char*)d_ws;
  size_t off = 0;
  auto carve = [&](size_t bytes) -> void* {
    void* p = ws + off;
    off = (off + bytes + 255) & ~(size_t)255;
    return p;
  };
  const size_t BF_NC = (size_t)NB * NN * CC * sizeof(__bf16);  // 4.72 MB
  __bf16* theta_t = (__bf16*)carve(BF_NC);
  __bf16* phi_t = (__bf16*)carve(BF_NC);
  __bf16* g_flat = (__bf16*)carve((size_t)NB * NI * HWp * sizeof(__bf16));
  __bf16* y2 = (__bf16*)carve((size_t)NB * HWp * NI * sizeof(__bf16));
  float* Mbuf = (float*)carve((size_t)NB * NN * sizeof(float));
  float* rZbuf = (float*)carve((size_t)NB * NN * sizeof(float));

  proj_kernel<<<dim3(HWp / 32, NB * 3), 256, 0, stream>>>(
      x1, x2, w_theta, w_phi, w_g, theta_t, phi_t, g_flat);
  stats_kernel<<<dim3(NN / 128, NB), 256, 0, stream>>>(theta_t, phi_t, Mbuf,
                                                       rZbuf);
  attn_kernel<<<dim3(NN / 128, NB), 256, 0, stream>>>(theta_t, phi_t, g_flat,
                                                      Mbuf, rZbuf, y2);
  mask_kernel<<<dim3(HWp / 16, NB), 256, 0, stream>>>(w_mask, y2, x1, out);
}
